// DisentangledSelfAttention_90537910599819
// MI455X (gfx1250) — matrix operand order repaired and checked by reading, not compiled, not run on hardware
//
#include <hip/hip_runtime.h>
#include <cmath>

typedef unsigned short u16;
typedef __bf16 bf16x8  __attribute__((ext_vector_type(8)));
typedef __bf16 bf16x16 __attribute__((ext_vector_type(16)));
typedef float  v8f     __attribute__((ext_vector_type(8)));
typedef u16    us8     __attribute__((ext_vector_type(8)));
typedef unsigned int u32x4 __attribute__((ext_vector_type(4)));
typedef int    i32x4   __attribute__((ext_vector_type(4)));
typedef int    i32x8   __attribute__((ext_vector_type(8)));

#define S_LEN 1024
#define CDIM  1024
#define NHEAD 16
#define DHEAD 64
#define SPAN_ 512

// ---------- helpers ----------
__device__ __forceinline__ u16 f2bf(float x) {
  union { float f; unsigned int u; } v; v.f = x;
  unsigned int r = v.u + 0x7FFFu + ((v.u >> 16) & 1u);  // round-nearest-even
  return (u16)(r >> 16);
}
__device__ __forceinline__ float bf2f(u16 h) {
  union { unsigned int u; float f; } v; v.u = ((unsigned int)h) << 16;
  return v.f;
}
__device__ __forceinline__ bf16x8 ld8(const u16* p) { return *(const bf16x8*)p; }
__device__ __forceinline__ bf16x16 join8(bf16x8 lo, bf16x8 hi) {
  return __builtin_shufflevector(lo, hi, 0,1,2,3,4,5,6,7,8,9,10,11,12,13,14,15);
}
__device__ __forceinline__ v8f wmma_bf16(bf16x16 a, bf16x16 b, v8f c) {
  return __builtin_amdgcn_wmma_f32_16x16x32_bf16(false, a, false, b, (short)0, c, false, false);
}
__device__ __forceinline__ int clampi(int x, int lo, int hi) {
  return x < lo ? lo : (x > hi ? hi : x);
}
// async 16B global->LDS copy (per-lane), tracked by ASYNCcnt
__device__ __forceinline__ void async_copy_b128(unsigned lds_off, const void* gptr) {
  unsigned long long ga = (unsigned long long)(uintptr_t)gptr;
  asm volatile("global_load_async_to_lds_b128 %0, %1, off" :: "v"(lds_off), "v"(ga) : "memory");
}
__device__ __forceinline__ void wait_asynccnt0() {
  asm volatile("s_wait_asynccnt 0" ::: "memory");
}

#if __has_builtin(__builtin_amdgcn_tensor_load_to_lds)
#define HAVE_TDM 1
// TDM: copy a 2D tile (rows x cols of bf16, row stride in elements) into LDS.
__device__ __forceinline__ void tdm_load_tile_2d(unsigned lds_addr, const void* gptr,
                                                 int cols, int rows, int row_stride) {
  unsigned long long ga = (unsigned long long)(uintptr_t)gptr;
  u32x4 g0;
  g0[0] = 1u;                                   // count=1, user descriptor
  g0[1] = lds_addr;                             // lds_addr [63:32]
  g0[2] = (unsigned)ga;                         // global_addr [95:64]
  g0[3] = (unsigned)(ga >> 32) | (2u << 30);    // global_addr hi + type=2 ("image")
  const unsigned td0 = 0x100000u;               // tensor_dim0 (large: no OOB clamp)
  const unsigned td1 = 0x100000u;               // tensor_dim1
  i32x8 g1;
  g1[0] = (int)(1u << 16);                                  // data_size=1 (2 bytes)
  g1[1] = (int)((td0 & 0xFFFFu) << 16);                     // tensor_dim0[15:0] at bits 63:48
  g1[2] = (int)((td0 >> 16) | ((td1 & 0xFFFFu) << 16));     // tensor_dim0 hi | tensor_dim1 lo
  g1[3] = (int)((td1 >> 16) | ((unsigned)cols << 16));      // tensor_dim1 hi | tile_dim0
  g1[4] = (int)((unsigned)rows);                            // tile_dim1 | tile_dim2=0
  g1[5] = (int)((unsigned)row_stride);                      // tensor_dim0_stride[31:0]
  g1[6] = 0;                                                // stride0 hi | stride1 lo
  g1[7] = 0;                                                // stride1 hi
  i32x4 g2 = {0, 0, 0, 0};
  i32x4 g3 = {0, 0, 0, 0};
#if defined(__clang_major__) && (__clang_major__ >= 23)
  i32x8 g4 = {0, 0, 0, 0, 0, 0, 0, 0};
  __builtin_amdgcn_tensor_load_to_lds(g0, g1, g2, g3, g4, 0);
#else
  __builtin_amdgcn_tensor_load_to_lds(g0, g1, g2, g3, 0);
#endif
}
#else
#define HAVE_TDM 0
#endif

// ---------- fp32 -> bf16 conversion ----------
__global__ void cvt_bf16_kernel(const float* __restrict__ in, u16* __restrict__ out, int n4) {
  int i = blockIdx.x * blockDim.x + threadIdx.x;
  if (i < n4) {
    float4 f = ((const float4*)in)[i];
    ushort4 o;
    o.x = f2bf(f.x); o.y = f2bf(f.y); o.z = f2bf(f.z); o.w = f2bf(f.w);
    ((ushort4*)out)[i] = o;
  }
}

// ---------- generic bf16 WMMA GEMM:  out = alpha * (A[MxK] @ W[KxN] + bias) ----------
// block: 256 threads (8 waves). Output tile: 128 (M) x 64 (N). K stepped by 32.
// W tile staged via GLOBAL_LOAD_ASYNC_TO_LDS_B128, then transposed in LDS.
template <bool OUT_F32, bool HAS_BIAS>
__global__ void __launch_bounds__(256) gemm_bf16_kernel(
    const u16* __restrict__ A, const u16* __restrict__ W,
    const float* __restrict__ bias, void* __restrict__ outp,
    int M, int N, int K, float alpha)
{
  __shared__ __align__(16) u16 wraw[32 * 64];   // W tile row-major (async landing zone)
  __shared__ __align__(16) u16 wt[64 * 32];     // W tile transposed: wt[n_local][k_local]
  const int tid  = threadIdx.x;
  const int wave = tid >> 5, lane = tid & 31;
  const int hf   = lane >> 4, lrow = lane & 15;
  const int m0 = blockIdx.x * 128 + wave * 16;
  const int n0 = blockIdx.y * 64;
  const unsigned wraw_base = (unsigned)(uintptr_t)(void*)wraw;

  v8f acc[4] = {};
  const u16* arow = A + (size_t)(m0 + lrow) * K;

  for (int k0 = 0; k0 < K; k0 += 32) {
    __syncthreads();
    {   // async-copy W[k0..k0+31][n0..n0+63] row-major into wraw (16B per thread)
      const u16* gsrc = W + (size_t)(k0 + (tid >> 3)) * N + n0 + (tid & 7) * 8;
      async_copy_b128(wraw_base + (unsigned)tid * 16u, gsrc);
      wait_asynccnt0();
    }
    __syncthreads();
    {   // transpose wraw -> wt inside LDS
      int kk = tid >> 3;            // 0..31
      int j0 = (tid & 7) * 8;       // 0..56
      us8 wv = *(const us8*)(wraw + kk * 64 + j0);
      #pragma unroll
      for (int i = 0; i < 8; ++i) wt[(j0 + i) * 32 + kk] = wv[i];
    }
    __syncthreads();

    // A tile (16x32): 16 consecutive K starting at 16*hf
    bf16x16 a = join8(ld8(arow + k0 + hf * 16), ld8(arow + k0 + hf * 16 + 8));
    __builtin_prefetch(arow + k0 + 32, 0, 1);
    #pragma unroll
    for (int nt = 0; nt < 4; ++nt) {
      // B tile (32x16): lane = column n, 16 consecutive K starting at 16*hf
      const u16* wr = wt + (nt * 16 + lrow) * 32;
      bf16x16 b = join8(ld8(wr + hf * 16), ld8(wr + hf * 16 + 8));
      acc[nt] = wmma_bf16(a, b, acc[nt]);
    }
  }

  #pragma unroll
  for (int nt = 0; nt < 4; ++nt) {
    int n = n0 + nt * 16 + lrow;
    float bsv = HAS_BIAS ? bias[n] : 0.0f;
    #pragma unroll
    for (int v = 0; v < 8; ++v) {
      int m = m0 + v + 8 * hf;
      float val = (acc[nt][v] + bsv) * alpha;
      if (OUT_F32) ((float*)outp)[(size_t)m * N + n] = val;
      else         ((u16*)outp)[(size_t)m * N + n] = f2bf(val);
    }
  }
}

// ---------- fused disentangled attention ----------
// grid: (B*H, S/64); block: 128 threads (4 waves). Each wave owns 16 q rows.
// dyn LDS: 4 waves * 16*1024 bf16 strips (128KB) + vt 64x32 (4KB) + vraw 32x64 (4KB).
__global__ void __launch_bounds__(128) attn_kernel(
    const u16* __restrict__ Q, const u16* __restrict__ Kb,
    const u16* __restrict__ V, const u16* __restrict__ PK,
    const u16* __restrict__ PQ, u16* __restrict__ O)
{
  extern __shared__ char smem[];
  u16* strip_all = (u16*)smem;                                  // [4][16][1024]
  u16* vt        = (u16*)(smem + 4 * 16 * S_LEN * 2);           // [64][32] transposed V
  u16* vraw      = (u16*)(smem + 4 * 16 * S_LEN * 2 + 64 * 32 * 2); // [32][64] TDM landing

  const int tid  = threadIdx.x;
  const int wave = tid >> 5, lane = tid & 31;
  const int hf   = lane >> 4, lrow = lane & 15;
  const int b = blockIdx.x >> 4, h = blockIdx.x & 15;
  const int q0 = blockIdx.y * 64 + wave * 16;
  u16* strip = strip_all + wave * 16 * S_LEN;

  // Load this wave's A tiles once: q rows (scaled) and k rows at the same positions
  const size_t abase = ((size_t)b * S_LEN + q0 + lrow) * CDIM + h * DHEAD;
  bf16x16 a_q[2], a_k[2];
  #pragma unroll
  for (int c = 0; c < 2; ++c) {
    a_q[c] = join8(ld8(Q  + abase + c * 32 + hf * 16), ld8(Q  + abase + c * 32 + hf * 16 + 8));
    a_k[c] = join8(ld8(Kb + abase + c * 32 + hf * 16), ld8(Kb + abase + c * 32 + hf * 16 + 8));
  }

  // ---- Pass 1: scores + c2p + p2c over all k tiles, store bf16 strip ----
  for (int j0 = 0; j0 < S_LEN; j0 += 16) {
    v8f s = {};
    {
      const size_t kb = ((size_t)b * S_LEN + j0 + lrow) * CDIM + h * DHEAD;
      #pragma unroll
      for (int c = 0; c < 2; ++c) {
        bf16x16 bk = join8(ld8(Kb + kb + c * 32 + hf * 16), ld8(Kb + kb + c * 32 + hf * 16 + 8));
        s = wmma_bf16(a_q[c], bk, s);
      }
    }
    // 16x32 mini-GEMMs over the 31-wide relative-position window
    v8f m1[2] = {};   // c2p: q . pos_key[clamp(q0+m - (j0+n) + 512)]
    v8f m2[2] = {};   // p2c: k . pos_query[clamp((j0+n) - (q0+m) + 512)]
    #pragma unroll
    for (int g = 0; g < 2; ++g) {
      int t  = g * 16 + lrow;
      int r1 = clampi(q0 - j0 + SPAN_ + t - 15, 0, 2 * SPAN_ - 1);
      int r2 = clampi(j0 - q0 + SPAN_ + t - 15, 0, 2 * SPAN_ - 1);
      const size_t p1 = (size_t)r1 * CDIM + h * DHEAD;
      const size_t p2 = (size_t)r2 * CDIM + h * DHEAD;
      #pragma unroll
      for (int c = 0; c < 2; ++c) {
        bf16x16 b1 = join8(ld8(PK + p1 + c * 32 + hf * 16), ld8(PK + p1 + c * 32 + hf * 16 + 8));
        m1[g] = wmma_bf16(a_q[c], b1, m1[g]);
        bf16x16 b2 = join8(ld8(PQ + p2 + c * 32 + hf * 16), ld8(PQ + p2 + c * 32 + hf * 16 + 8));
        m2[g] = wmma_bf16(a_k[c], b2, m2[g]);
      }
    }
    // anti-diagonal gather from the minis via cross-lane shuffles
    #pragma unroll
    for (int v = 0; v < 8; ++v) {
      int m = v + 8 * hf;
      int n = lrow;
      int t1 = 15 + m - n;          // 0..30
      int t2 = 15 + n - m;          // 0..30
      int sl1 = (t1 & 15) + (hf << 4);
      int sl2 = (t2 & 15) + (hf << 4);
      float g1a = __shfl(m1[0][v], sl1, 32);
      float g1b = __shfl(m1[1][v], sl1, 32);
      float g2a = __shfl(m2[0][v], sl2, 32);
      float g2b = __shfl(m2[1][v], sl2, 32);
      float g1 = (t1 < 16) ? g1a : g1b;
      float g2 = (t2 < 16) ? g2a : g2b;
      strip[m * S_LEN + j0 + n] = f2bf(s[v] + g1 + g2);
    }
  }

  // ---- Pass 2: row softmax in the strip (lane pair L / L+16 splits columns) ----
  float mx = -3.0e38f;
  u16* srow = strip + lrow * S_LEN + hf * 512;
  for (int c0 = 0; c0 < 512; c0 += 8) {
    us8 vv = *(const us8*)(srow + c0);
    #pragma unroll
    for (int i = 0; i < 8; ++i) mx = fmaxf(mx, bf2f(vv[i]));
  }
  mx = fmaxf(mx, __shfl_xor(mx, 16, 32));
  float sum = 0.0f;
  for (int c0 = 0; c0 < 512; c0 += 8) {
    us8 vv = *(const us8*)(srow + c0);
    us8 ov;
    #pragma unroll
    for (int i = 0; i < 8; ++i) {
      float p = __expf(bf2f(vv[i]) - mx);
      sum += p;
      ov[i] = f2bf(p);
    }
    *(us8*)(srow + c0) = ov;
  }
  sum += __shfl_xor(sum, 16, 32);

  // ---- Pass 3: P @ V ;  V tile staged by the Tensor Data Mover ----
  v8f o[4] = {};
  for (int kc = 0; kc < S_LEN; kc += 32) {
    __syncthreads();
#if HAVE_TDM
    if (wave == 0) {
      // TDM: 2D tile, 64 bf16 per row (d), 32 rows (kpos), row stride CDIM elements
      tdm_load_tile_2d((unsigned)(uintptr_t)(void*)vraw,
                       V + ((size_t)b * S_LEN + kc) * CDIM + h * DHEAD,
                       /*cols=*/DHEAD, /*rows=*/32, /*row_stride=*/CDIM);
    }
    __builtin_amdgcn_s_wait_tensorcnt(0);
    __syncthreads();
    {   // transpose vraw[kpos][d] -> vt[d][kpos]
      int kp = tid >> 2;            // 0..31
      int d0 = (tid & 3) * 16;      // 0,16,32,48
      const u16* vp = vraw + kp * 64 + d0;
      us8 v0 = *(const us8*)vp;
      us8 v1 = *(const us8*)(vp + 8);
      #pragma unroll
      for (int i = 0; i < 8; ++i) {
        vt[(d0 + i) * 32 + kp]     = v0[i];
        vt[(d0 + 8 + i) * 32 + kp] = v1[i];
      }
    }
#else
    {   // fallback: cooperative global loads + transpose
      int kp = tid >> 2;
      int d0 = (tid & 3) * 16;
      const u16* vp = V + ((size_t)b * S_LEN + kc + kp) * CDIM + h * DHEAD + d0;
      us8 v0 = *(const us8*)vp;
      us8 v1 = *(const us8*)(vp + 8);
      #pragma unroll
      for (int i = 0; i < 8; ++i) {
        vt[(d0 + i) * 32 + kp]     = v0[i];
        vt[(d0 + 8 + i) * 32 + kp] = v1[i];
      }
    }
#endif
    __syncthreads();
    const u16* pr = strip + lrow * S_LEN + kc;
    bf16x16 a_p = join8(ld8(pr + hf * 16), ld8(pr + hf * 16 + 8));
    #pragma unroll
    for (int dt = 0; dt < 4; ++dt) {
      const u16* vr = vt + (dt * 16 + lrow) * 32;
      bf16x16 bv = join8(ld8(vr + hf * 16), ld8(vr + hf * 16 + 8));
      o[dt] = wmma_bf16(a_p, bv, o[dt]);
    }
  }

  // ---- normalize and write attn_out (bf16) ----
  #pragma unroll
  for (int v = 0; v < 8; ++v) {
    int m = v + 8 * hf;
    float inv = 1.0f / __shfl(sum, m, 32);
    #pragma unroll
    for (int dt = 0; dt < 4; ++dt) {
      int d = dt * 16 + lrow;
      O[((size_t)b * S_LEN + q0 + m) * CDIM + h * DHEAD + d] = f2bf(o[dt][v] * inv);
    }
  }
}

// ---------- host launch ----------
extern "C" void kernel_launch(void* const* d_in, const int* in_sizes, int n_in,
                              void* d_out, int out_size, void* d_ws, size_t ws_size,
                              hipStream_t stream) {
  (void)in_sizes; (void)n_in; (void)out_size; (void)ws_size;
  const float* hs   = (const float*)d_in[0];
  const float* rel  = (const float*)d_in[1];
  const float* Wq   = (const float*)d_in[2];
  const float* bq   = (const float*)d_in[3];
  const float* Wk   = (const float*)d_in[4];
  const float* Wv   = (const float*)d_in[5];
  const float* bv   = (const float*)d_in[6];
  const float* Wc2p = (const float*)d_in[7];
  const float* Wp2c = (const float*)d_in[8];
  const float* bp2c = (const float*)d_in[9];
  const float* Wo   = (const float*)d_in[10];
  const float* bo   = (const float*)d_in[11];
  float* out = (float*)d_out;

  const size_t BS = 4 * 1024;          // B*S rows
  const size_t MC = 1024 * 1024;       // 1M elements
  u16* p = (u16*)d_ws;
  u16* hs_bf  = p; p += BS * 1024;
  u16* rel_bf = p; p += MC;
  u16* wq_bf  = p; p += MC;
  u16* wk_bf  = p; p += MC;
  u16* wv_bf  = p; p += MC;
  u16* wc_bf  = p; p += MC;            // Wc2p
  u16* wp_bf  = p; p += MC;            // Wp2c
  u16* wo_bf  = p; p += MC;
  u16* q_bf   = p; p += BS * 1024;
  u16* k_bf   = p; p += BS * 1024;
  u16* v_bf   = p; p += BS * 1024;
  u16* pk_bf  = p; p += MC;            // pos_key
  u16* pq_bf  = p; p += MC;            // pos_query
  u16* ao_bf  = p; p += BS * 1024;     // attention output

  const float scale = 1.0f / sqrtf(3.0f * 64.0f);

  auto cvt = [&](const float* src, u16* dst, int n) {
    int n4 = n / 4;
    cvt_bf16_kernel<<<(n4 + 255) / 256, 256, 0, stream>>>(src, dst, n4);
  };
  cvt(hs,   hs_bf,  (int)(BS * 1024));
  cvt(rel,  rel_bf, (int)MC);
  cvt(Wq,   wq_bf,  (int)MC);
  cvt(Wk,   wk_bf,  (int)MC);
  cvt(Wv,   wv_bf,  (int)MC);
  cvt(Wc2p, wc_bf,  (int)MC);
  cvt(Wp2c, wp_bf,  (int)MC);
  cvt(Wo,   wo_bf,  (int)MC);

  dim3 gProj(4096 / 128, 1024 / 64);
  dim3 gRel(1024 / 128, 1024 / 64);
  gemm_bf16_kernel<false, true ><<<gProj, 256, 0, stream>>>(hs_bf, wq_bf, bq, q_bf, 4096, 1024, 1024, scale);
  gemm_bf16_kernel<false, false><<<gProj, 256, 0, stream>>>(hs_bf, wk_bf, nullptr, k_bf, 4096, 1024, 1024, 1.0f);
  gemm_bf16_kernel<false, true ><<<gProj, 256, 0, stream>>>(hs_bf, wv_bf, bv, v_bf, 4096, 1024, 1024, 1.0f);
  gemm_bf16_kernel<false, false><<<gRel, 256, 0, stream>>>(rel_bf, wc_bf, nullptr, pk_bf, 1024, 1024, 1024, 1.0f);
  gemm_bf16_kernel<false, true ><<<gRel, 256, 0, stream>>>(rel_bf, wp_bf, bp2c, pq_bf, 1024, 1024, 1024, scale);

  const int attn_lds = 4 * 16 * 1024 * 2 + 64 * 32 * 2 + 32 * 64 * 2;
  attn_kernel<<<dim3(64, 16), 128, attn_lds, stream>>>(q_bf, k_bf, v_bf, pk_bf, pq_bf, ao_bf);

  gemm_bf16_kernel<true, true><<<gProj, 256, 0, stream>>>(ao_bf, wo_bf, bo, out, 4096, 1024, 1024, 1.0f);
}
